// Attention_KV_1726576853355
// MI455X (gfx1250) — matrix operand order repaired and checked by reading, not compiled, not run on hardware
//
#include <hip/hip_runtime.h>
#include <hip/hip_bf16.h>

typedef __attribute__((ext_vector_type(16))) _Float16 v16h;
typedef __attribute__((ext_vector_type(8)))  float    v8f;
typedef unsigned int uint;
typedef __attribute__((ext_vector_type(4))) uint u32x4;
typedef __attribute__((ext_vector_type(8))) int  i32x8;
typedef __attribute__((ext_vector_type(4))) int  i32x4;

#define Bq 8
#define Nq 1024
#define DIMq 512
#define Hq 8
#define Dq 64

// ---------------- workspace layout (bytes) ----------------
#define OFF_XH    0                         // 8192*512 f16   = 8 MiB
#define OFF_WKVH  (8388608)                 // 1024*512 f16   = 1 MiB
#define OFF_WOUTH (OFF_WKVH + 1048576)      // 512*512 f16
#define OFF_KH    (OFF_WOUTH + 524288)      // (b,h,n,d) f16  = 8 MiB
#define OFF_VH    (OFF_KH + 8388608)        // (b,h,n,d) f16  = 8 MiB
#define OFF_AOH   (OFF_VH + 8388608)        // (b*n, dim) f16 = 8 MiB
#define OFF_PB    (OFF_AOH + 8388608)       // (n,n) f32      = 4 MiB

// ---------------- WMMA helpers ----------------
__device__ __forceinline__ v8f wmma16(v16h a, v16h b, v8f c) {
  return __builtin_amdgcn_wmma_f32_16x16x32_f16(false, a, false, b, (short)0, c,
                                                false, false);
}

// A-fragment gather: 16x32 f16, lane = M, K contiguous per half-wave.
__device__ __forceinline__ v16h gatherA(const uint* rowp, int g) {
  union { v16h h; uint4 q[2]; } r;
  r.q[0] = *(const uint4*)(rowp + 8 * g);
  r.q[1] = *(const uint4*)(rowp + 8 * g + 4);
  return r.h;
}

// B-fragment gather: 32x16 f16, lane = N, K contiguous per half-wave.
__device__ __forceinline__ v16h gatherB(const uint* rowp, int g) {
  union { v16h h; uint4 q[2]; } r;
  r.q[0] = *(const uint4*)(rowp + 8 * g);
  r.q[1] = *(const uint4*)(rowp + 8 * g + 4);
  return r.h;
}

// B-fragment gather from a ROW-MAJOR [j][d] tile (transposed access):
// lane = N = column dcol; element pair v -> rows j = 16g+2v, 16g+2v+1.
__device__ __forceinline__ v16h gatherBT(const _Float16* tile, int strideHalfs,
                                         int dcol, int g) {
  union { v16h h; uint u[8]; } r;
  const unsigned short* t = (const unsigned short*)tile;
  #pragma unroll
  for (int v = 0; v < 8; ++v) {
    int j = 16 * g + 2 * v;
    uint lo = t[j * strideHalfs + dcol];
    uint hi = t[(j + 1) * strideHalfs + dcol];
    r.u[v] = lo | (hi << 16);
  }
  return r.h;
}

__device__ __forceinline__ float rmax16(float x) {
  #pragma unroll
  for (int m = 8; m >= 1; m >>= 1) x = fmaxf(x, __shfl_xor(x, m, 16));
  return x;
}
__device__ __forceinline__ float rsum16(float x) {
  #pragma unroll
  for (int m = 8; m >= 1; m >>= 1) x += __shfl_xor(x, m, 16);
  return x;
}

// ---------------- Tensor Data Mover: 2-D f16 tile -> LDS ----------------
__device__ __forceinline__ void tdm_load_2d(const void* gsrc, uint lds_off,
                                            uint width_elems, uint rows,
                                            uint row_stride_elems,
                                            uint pad_interval, uint pad_amount) {
  unsigned long long ga = (unsigned long long)(size_t)gsrc;
  uint lo = (uint)__builtin_amdgcn_readfirstlane((int)(ga & 0xffffffffu));
  uint hi = (uint)__builtin_amdgcn_readfirstlane((int)(ga >> 32));
  uint ld = (uint)__builtin_amdgcn_readfirstlane((int)lds_off);
  u32x4 g0 = {1u,                                  // count=1, no gather
              ld,                                  // lds_addr (bytes)
              lo,                                  // global_addr[31:0]
              (hi & 0x01ffffffu) | (2u << 30)};    // global_addr[56:32] | type=2
  i32x8 g1 = {(int)((1u << 16)                     // data_size = 2 bytes
                    | (1u << 20)                   // pad_enable
                    | (pad_interval << 22) | (pad_amount << 25)),
              (int)(width_elems << 16),            // tensor_dim0[15:0]
              (int)(rows << 16),                   // tensor_dim1[15:0]
              (int)(width_elems << 16),            // tile_dim0
              (int)rows,                           // tile_dim1 (tile_dim2=0)
              (int)row_stride_elems,               // tensor_dim0_stride[31:0]
              0, 0};
  i32x4 z4 = {0, 0, 0, 0};
#if __has_include(<hip/amd_detail/amd_gfx1250_TDM.h>)
  i32x8 z8 = {0, 0, 0, 0, 0, 0, 0, 0};
  __builtin_amdgcn_tensor_load_to_lds(g0, g1, z4, z4, z8, 0);
#else
  __builtin_amdgcn_tensor_load_to_lds(g0, g1, z4, z4, 0);
#endif
}

// ---------------- kernel 1: f32 -> f16 conversions ----------------
#define NX  (Bq * Nq * DIMq)        // 4194304
#define NKV (2 * DIMq * DIMq)       // 524288
#define NWO (DIMq * DIMq)           // 262144
__global__ void cvt_kernel(const float* __restrict__ x,
                           const float* __restrict__ wkv,
                           const float* __restrict__ wout,
                           _Float16* __restrict__ xh,
                           _Float16* __restrict__ wkvh,
                           _Float16* __restrict__ wouth) {
  size_t i = (size_t)blockIdx.x * 256 + threadIdx.x;
  if (i < NX)                  xh[i]               = (_Float16)x[i];
  else if (i < NX + NKV)       wkvh[i - NX]        = (_Float16)wkv[i - NX];
  else if (i < NX + NKV + NWO) wouth[i - NX - NKV] = (_Float16)wout[i - NX - NKV];
}

// ---------------- kernel 2: pos_bias = pos @ w_pos + b_pos ----------------
__global__ void posbias_kernel(const float* __restrict__ pos,
                               const float* __restrict__ w_pos,
                               const float* __restrict__ b_pos,
                               float* __restrict__ pb) {
  size_t idx = (size_t)blockIdx.x * 256 + threadIdx.x;   // 0..N*N-1
  const float2* p = (const float2*)(pos + idx * 50);
  float acc = b_pos[0];
  #pragma unroll
  for (int i = 0; i < 25; ++i) {
    float2 t = p[i];
    acc += t.x * w_pos[2 * i] + t.y * w_pos[2 * i + 1];
  }
  pb[idx] = acc;
}

// ---------------- TDM + double-buffered WMMA GEMM core ----------------
#define G_ASTR 24   // LDS row stride in uints (48 halfs; b128-aligned)
#define AT_BUF (128 * G_ASTR)
#define BT_BUF (64 * G_ASTR)

// ---------------- kernel 3: kv = x @ Wkv^T, scatter to K/V (b,h,n,d) --------
__global__ __launch_bounds__(256) void kvgemm_kernel(
    const _Float16* __restrict__ xh, const _Float16* __restrict__ wkvh,
    _Float16* __restrict__ kh, _Float16* __restrict__ vh) {
  __shared__ __align__(16) uint At[2 * AT_BUF];
  __shared__ __align__(16) uint Bt[2 * BT_BUF];
  const int tid = threadIdx.x, lane = tid & 31, w = tid >> 5;
  const int g = lane >> 4, l15 = lane & 15;
  const int n0 = blockIdx.x * 64;
  const size_t m0 = (size_t)blockIdx.y * 128;
  const uint ldsA = (uint)(size_t)At, ldsB = (uint)(size_t)Bt;

  v8f C0 = {0,0,0,0,0,0,0,0}, C1 = C0, C2 = C0, C3 = C0;

  if (w == 0) {   // prefetch kk=0 tiles via TDM (rows 64B, pad to 96B stride)
    tdm_load_2d(xh + m0 * DIMq, ldsA, 32, 128, DIMq, 3, 7);
    tdm_load_2d(wkvh + (size_t)n0 * DIMq, ldsB, 32, 64, DIMq, 3, 7);
  }
  for (int kk = 0; kk < DIMq; kk += 32) {
    const int buf = (kk >> 5) & 1;
    if (w == 0) __builtin_amdgcn_s_wait_tensorcnt(0);
    __syncthreads();
    if (w == 0 && kk + 32 < DIMq) {
      tdm_load_2d(xh + m0 * DIMq + kk + 32, ldsA + (buf ^ 1) * (AT_BUF * 4),
                  32, 128, DIMq, 3, 7);
      tdm_load_2d(wkvh + (size_t)n0 * DIMq + kk + 32,
                  ldsB + (buf ^ 1) * (BT_BUF * 4), 32, 64, DIMq, 3, 7);
    }
    const uint* at = At + buf * AT_BUF;
    const uint* bt = Bt + buf * BT_BUF;
    // hoist all fragment loads so DS reads pipeline ahead of the WMMA group
    v16h a  = gatherA(at + (w * 16 + l15) * G_ASTR, g);
    v16h b0 = gatherB(bt + (l15) * G_ASTR, g);
    v16h b1 = gatherB(bt + (16 + l15) * G_ASTR, g);
    v16h b2 = gatherB(bt + (32 + l15) * G_ASTR, g);
    v16h b3 = gatherB(bt + (48 + l15) * G_ASTR, g);
    C0 = wmma16(a, b0, C0);
    C1 = wmma16(a, b1, C1);
    C2 = wmma16(a, b2, C2);
    C3 = wmma16(a, b3, C3);
  }
  #pragma unroll
  for (int v = 0; v < 8; ++v) {
    size_t mrow = m0 + w * 16 + v + 8 * g;
    int b = (int)(mrow >> 10), n = (int)(mrow & 1023);
    #pragma unroll
    for (int nc = 0; nc < 4; ++nc) {
      int o = n0 + nc * 16 + l15;
      float val = (nc == 0 ? C0[v] : nc == 1 ? C1[v] : nc == 2 ? C2[v] : C3[v]);
      int oo = o & 511, h = oo >> 6, d = oo & 63;
      _Float16* dst = (o < DIMq) ? kh : vh;
      dst[(((size_t)b * Hq + h) * Nq + n) * Dq + d] = (_Float16)val;
    }
  }
}

// ---------------- kernel 4: flash attention, TDM-fed, f16 WMMA --------------
#define AT_KSTR 40            // K tile row stride (uints): 128B row + 32B pad
#define AT_VSTR 41            // V tile row stride (uints): odd -> bank spread
#define AT_PSTR 24
#define KT_BUF (32 * AT_KSTR)
#define VT_BUF (32 * AT_VSTR)
__global__ __launch_bounds__(256) void attn_kernel(
    const _Float16* __restrict__ kh, const _Float16* __restrict__ vh,
    const float* __restrict__ pb, const float* __restrict__ w_pos,
    _Float16* __restrict__ aoh) {
  __shared__ __align__(16) uint Kt[2 * KT_BUF];
  __shared__ __align__(16) uint Vt[2 * VT_BUF];
  __shared__ __align__(16) uint Pt[8 * 16 * AT_PSTR];
  const int tid = threadIdx.x, lane = tid & 31, w = tid >> 5;
  const int g = lane >> 4, l15 = lane & 15;
  const int bh = blockIdx.x;
  const int r0 = blockIdx.y * 128 + w * 16;
  const uint ldsK = (uint)(size_t)Kt, ldsV = (uint)(size_t)Vt;
  const size_t bhbase = (size_t)bh * Nq * Dq;

  float sumw = 0.f;
  for (int p = 0; p < 50; ++p) sumw += w_pos[p];
  const float sw = sumw * 0.04419417382415922f;        // * dim^-0.5

  // loop-invariant A fragments: this wave's 16 K-rows (16x64)
  const uint* rp = (const uint*)kh + (bhbase >> 1) + (size_t)(r0 + l15) * (Dq / 2);
  v16h a0 = gatherA(rp, g);
  v16h a1 = gatherA(rp + 16, g);

  v8f O0 = {0,0,0,0,0,0,0,0}, O1 = O0, O2 = O0, O3 = O0;
  float m[8], l[8];
  #pragma unroll
  for (int v = 0; v < 8; ++v) { m[v] = -3e38f; l[v] = 0.f; }

  if (w == 0) {   // prefetch first K/V tiles
    tdm_load_2d(kh + bhbase, ldsK, 64, 32, 64, 4, 7);
    tdm_load_2d(vh + bhbase, ldsV, 64, 32, 64, 4, 8);
  }
  for (int jb = 0; jb < Nq / 32; ++jb) {
    const int buf = jb & 1;
    if (w == 0) __builtin_amdgcn_s_wait_tensorcnt(0);
    __syncthreads();
    if (w == 0 && jb + 1 < Nq / 32) {
      size_t nsrc = bhbase + (size_t)(jb + 1) * 32 * Dq;
      tdm_load_2d(kh + nsrc, ldsK + (buf ^ 1) * (KT_BUF * 4), 64, 32, 64, 4, 7);
      tdm_load_2d(vh + nsrc, ldsV + (buf ^ 1) * (VT_BUF * 4), 64, 32, 64, 4, 8);
    }
    const uint* ktb = Kt + buf * KT_BUF;
    const _Float16* vtb = (const _Float16*)(Vt + buf * VT_BUF);

    // S = Krows @ Ktile^T : hoist the 4 B fragments, then 4 WMMAs
    v16h bk0 = gatherB(ktb + l15 * AT_KSTR, g);
    v16h bk1 = gatherB(ktb + l15 * AT_KSTR + 16, g);
    v16h bk2 = gatherB(ktb + (16 + l15) * AT_KSTR, g);
    v16h bk3 = gatherB(ktb + (16 + l15) * AT_KSTR + 16, g);
    v8f S0 = {0,0,0,0,0,0,0,0}, S1 = S0;
    S0 = wmma16(a0, bk0, S0);
    S0 = wmma16(a1, bk1, S0);
    S1 = wmma16(a0, bk2, S1);
    S1 = wmma16(a1, bk3, S1);

    // scores + online softmax (row = v + 8g, cols = lanes)
    const int j0 = jb * 32;
    float scl[8], p0[8], p1[8];
    #pragma unroll
    for (int v = 0; v < 8; ++v) {
      int i = r0 + v + 8 * g;
      float s0 = S0[v] * sw + pb[(size_t)i * Nq + j0 + l15];
      float s1 = S1[v] * sw + pb[(size_t)i * Nq + j0 + 16 + l15];
      float mt = rmax16(fmaxf(s0, s1));
      float mn = fmaxf(m[v], mt);
      float sc = __expf(m[v] - mn);
      float e0 = __expf(s0 - mn), e1 = __expf(s1 - mn);
      l[v] = l[v] * sc + rsum16(e0 + e1);
      m[v] = mn; scl[v] = sc; p0[v] = e0; p1[v] = e1;
    }
    #pragma unroll
    for (int v = 0; v < 8; ++v) {
      O0[v] *= scl[v]; O1[v] *= scl[v]; O2[v] *= scl[v]; O3[v] *= scl[v];
    }

    // stage P (C-layout -> A-layout) through per-wave LDS
    _Float16* pst = (_Float16*)(Pt + w * 16 * AT_PSTR);
    #pragma unroll
    for (int v = 0; v < 8; ++v) {
      int M = v + 8 * g;
      pst[M * (2 * AT_PSTR) + l15]      = (_Float16)p0[v];
      pst[M * (2 * AT_PSTR) + 16 + l15] = (_Float16)p1[v];
    }
    asm volatile("s_wait_dscnt 0" ::: "memory");
    v16h pa = gatherA(Pt + (w * 16 + l15) * AT_PSTR, g);

    // O += P @ Vtile : hoist the 4 V fragments, then 4 WMMAs
    v16h bv0 = gatherBT(vtb, 2 * AT_VSTR, l15, g);
    v16h bv1 = gatherBT(vtb, 2 * AT_VSTR, 16 + l15, g);
    v16h bv2 = gatherBT(vtb, 2 * AT_VSTR, 32 + l15, g);
    v16h bv3 = gatherBT(vtb, 2 * AT_VSTR, 48 + l15, g);
    O0 = wmma16(pa, bv0, O0);
    O1 = wmma16(pa, bv1, O1);
    O2 = wmma16(pa, bv2, O2);
    O3 = wmma16(pa, bv3, O3);
  }

  // epilogue: O /= l, store f16 to attn_out[(b*N+n), h*64 + d]
  const int b = bh >> 3, h = bh & 7;
  #pragma unroll
  for (int v = 0; v < 8; ++v) {
    float inv = 1.0f / l[v];
    int n = r0 + v + 8 * g;
    size_t rowb = ((size_t)b * Nq + n) * DIMq + h * Dq;
    aoh[rowb + l15]      = (_Float16)(O0[v] * inv);
    aoh[rowb + 16 + l15] = (_Float16)(O1[v] * inv);
    aoh[rowb + 32 + l15] = (_Float16)(O2[v] * inv);
    aoh[rowb + 48 + l15] = (_Float16)(O3[v] * inv);
  }
}

// ---------------- kernel 5: out = attn_out @ Wout^T + b_out ----------------
__global__ __launch_bounds__(256) void outproj_kernel(
    const _Float16* __restrict__ ah, const _Float16* __restrict__ wouth,
    const float* __restrict__ bout, float* __restrict__ out) {
  __shared__ __align__(16) uint At[2 * AT_BUF];
  __shared__ __align__(16) uint Bt[2 * BT_BUF];
  const int tid = threadIdx.x, lane = tid & 31, w = tid >> 5;
  const int g = lane >> 4, l15 = lane & 15;
  const int n0 = blockIdx.x * 64;
  const size_t m0 = (size_t)blockIdx.y * 128;
  const uint ldsA = (uint)(size_t)At, ldsB = (uint)(size_t)Bt;

  v8f C0 = {0,0,0,0,0,0,0,0}, C1 = C0, C2 = C0, C3 = C0;
  if (w == 0) {
    tdm_load_2d(ah + m0 * DIMq, ldsA, 32, 128, DIMq, 3, 7);
    tdm_load_2d(wouth + (size_t)n0 * DIMq, ldsB, 32, 64, DIMq, 3, 7);
  }
  for (int kk = 0; kk < DIMq; kk += 32) {
    const int buf = (kk >> 5) & 1;
    if (w == 0) __builtin_amdgcn_s_wait_tensorcnt(0);
    __syncthreads();
    if (w == 0 && kk + 32 < DIMq) {
      tdm_load_2d(ah + m0 * DIMq + kk + 32, ldsA + (buf ^ 1) * (AT_BUF * 4),
                  32, 128, DIMq, 3, 7);
      tdm_load_2d(wouth + (size_t)n0 * DIMq + kk + 32,
                  ldsB + (buf ^ 1) * (BT_BUF * 4), 32, 64, DIMq, 3, 7);
    }
    const uint* at = At + buf * AT_BUF;
    const uint* bt = Bt + buf * BT_BUF;
    v16h a  = gatherA(at + (w * 16 + l15) * G_ASTR, g);
    v16h b0 = gatherB(bt + (l15) * G_ASTR, g);
    v16h b1 = gatherB(bt + (16 + l15) * G_ASTR, g);
    v16h b2 = gatherB(bt + (32 + l15) * G_ASTR, g);
    v16h b3 = gatherB(bt + (48 + l15) * G_ASTR, g);
    C0 = wmma16(a, b0, C0);
    C1 = wmma16(a, b1, C1);
    C2 = wmma16(a, b2, C2);
    C3 = wmma16(a, b3, C3);
  }
  #pragma unroll
  for (int v = 0; v < 8; ++v) {
    size_t mrow = m0 + w * 16 + v + 8 * g;
    #pragma unroll
    for (int nc = 0; nc < 4; ++nc) {
      int o = n0 + nc * 16 + l15;
      float val = (nc == 0 ? C0[v] : nc == 1 ? C1[v] : nc == 2 ? C2[v] : C3[v]);
      out[mrow * DIMq + o] = val + bout[o];
    }
  }
}

// ---------------- launcher ----------------
extern "C" void kernel_launch(void* const* d_in, const int* in_sizes, int n_in,
                              void* d_out, int out_size, void* d_ws, size_t ws_size,
                              hipStream_t stream) {
  const float* x     = (const float*)d_in[0];
  const float* pos   = (const float*)d_in[1];
  const float* W_kv  = (const float*)d_in[2];
  const float* W_out = (const float*)d_in[3];
  const float* b_out = (const float*)d_in[4];
  const float* w_pos = (const float*)d_in[5];
  const float* b_pos = (const float*)d_in[6];

  char* ws = (char*)d_ws;
  _Float16* xh    = (_Float16*)(ws + OFF_XH);
  _Float16* wkvh  = (_Float16*)(ws + OFF_WKVH);
  _Float16* wouth = (_Float16*)(ws + OFF_WOUTH);
  _Float16* kh    = (_Float16*)(ws + OFF_KH);
  _Float16* vh    = (_Float16*)(ws + OFF_VH);
  _Float16* aoh   = (_Float16*)(ws + OFF_AOH);
  float*    pb    = (float*)(ws + OFF_PB);

  cvt_kernel<<<(NX + NKV + NWO) / 256, 256, 0, stream>>>(x, W_kv, W_out, xh, wkvh, wouth);
  posbias_kernel<<<(Nq * Nq) / 256, 256, 0, stream>>>(pos, w_pos, b_pos, pb);
  kvgemm_kernel<<<dim3(2 * DIMq / 64, Bq * Nq / 128), 256, 0, stream>>>(xh, wkvh, kh, vh);
  attn_kernel<<<dim3(Bq * Hq, Nq / 128), 256, 0, stream>>>(kh, vh, pb, w_pos, aoh);
  outproj_kernel<<<dim3(DIMq / 64, Bq * Nq / 128), 256, 0, stream>>>(aoh, wouth, b_out, (float*)d_out);
}
